// Net_GCN_WoP_WoEigen_56727928046292
// MI455X (gfx1250) — matrix operand order repaired and checked by reading, not compiled, not run on hardware
//
#include <hip/hip_runtime.h>
#include <hip/hip_bf16.h>

// ---------------------------------------------------------------------------
// GCN (3 blocks x 3 GCNConv + BN) -> mean pool -> FC -> FC for MI455X/gfx1250.
// Dense GEMMs use v_wmma_f32_16x16x32_bf16 (wave32 WMMA, f32 accumulate).
// Layer-0 input is zero-padded 19 -> 32 so every GEMM K-dim is a multiple of
// 32: the inner loop is branch-free b128 loads + 4 WMMAs (16x64 tile / wave).
// ---------------------------------------------------------------------------

#define NNODES 100000
#define NEDGES 1000000
#define NGRAPH 64
#define NCLASS 40
#define BN_EPS 1e-5f

typedef __bf16 bf16;
typedef __attribute__((ext_vector_type(16))) __bf16 v16bf;
typedef __attribute__((ext_vector_type(8)))  __bf16 bf16x8;
typedef __attribute__((ext_vector_type(8)))  float  v8f;

// ------------- feature mask: x[:, 0:6]++x[:, 26:39] -> bf16, pad to 32 -----
__global__ void feat_mask_kernel(const float* __restrict__ x, bf16* __restrict__ h, int n) {
  int t = blockIdx.x * blockDim.x + threadIdx.x;
  if (t >= n * 32) return;
  int node = t >> 5, c = t & 31;
  float v = 0.f;
  if (c < 19) {
    int sc = (c < 6) ? c : (c + 20);            // 6 -> 26 ... 18 -> 38
    v = x[(size_t)node * 39 + sc];
  }
  h[t] = (bf16)v;
}

// ------------- W [fi,fo] f32 -> Wt [fo,fi_pad] bf16 (zero-padded K) ---------
__global__ void conv_w_kernel(const float* __restrict__ W, bf16* __restrict__ Wt,
                              int fi, int fi_pad, int fo) {
  int t = blockIdx.x * blockDim.x + threadIdx.x;
  if (t >= fi_pad * fo) return;
  int j = t / fi_pad, k = t % fi_pad;
  Wt[t] = (k < fi) ? (bf16)W[(size_t)k * fo + j] : (bf16)0.f;
}

// ------------- GEMM: C[n,fo] = A[n,fi](bf16) @ Wt[fo,fi](bf16) --------------
// One wave per 16(M) x 64(N) output tile: A fragment reused across 4 WMMAs.
// A frag (16x32 bf16): lane L (g=L>>4, lr=L&15) row=lr, K = k0 + 8g + (i&7) + 16*(i>>3)
// B frag (32x16 bf16): lane L col=lr,            K = k0 + 8g + (i&7) + 16*(i>>3)
// C frag (16x16 f32):  element e -> row = e + 8g, col = lr
__global__ void gemm_bf16_wmma_kernel(const bf16* __restrict__ A,
                                      const bf16* __restrict__ Wt,
                                      float* __restrict__ C,
                                      int nrows, int fi, int fo) {
  int rtile = blockIdx.x * blockDim.y + threadIdx.y;
  int cbase = blockIdx.y * 4;                   // 4 ctiles = 64 output cols
  if (rtile * 16 >= nrows) return;              // wave-uniform exit
  int lane = threadIdx.x;
  int g = lane >> 4, lr = lane & 15;

  const bf16* Arow = A + (size_t)(rtile * 16 + lr) * fi;
  const bf16* Bbase = Wt + (size_t)(cbase * 16 + lr) * fi + 8 * g;

  v8f acc[4] = {{}, {}, {}, {}};
  for (int k0 = 0; k0 < fi; k0 += 32) {
    bf16x8 a0 = *(const bf16x8*)(Arow + k0 + 8 * g);
    bf16x8 a1 = *(const bf16x8*)(Arow + k0 + 16 + 8 * g);
    v16bf a;
#pragma unroll
    for (int i = 0; i < 8; ++i) { a[i] = a0[i]; a[i + 8] = a1[i]; }

#pragma unroll
    for (int c = 0; c < 4; ++c) {
      const bf16* Bcol = Bbase + (size_t)c * 16 * fi + k0;
      bf16x8 b0 = *(const bf16x8*)(Bcol);
      bf16x8 b1 = *(const bf16x8*)(Bcol + 16);
      v16bf b;
#pragma unroll
      for (int i = 0; i < 8; ++i) { b[i] = b0[i]; b[i + 8] = b1[i]; }
      acc[c] = __builtin_amdgcn_wmma_f32_16x16x32_bf16(
          false, a, false, b, (short)0, acc[c], false, false);
    }
  }

#pragma unroll
  for (int c = 0; c < 4; ++c) {
    float* Cp = C + (size_t)(rtile * 16 + 8 * g) * fo + (cbase + c) * 16 + lr;
#pragma unroll
    for (int e = 0; e < 8; ++e) {
      Cp[(size_t)e * fo] = acc[c][e];
    }
  }
}

// ------------- edge scatter: out[dst] += msg[src] ---------------------------
__global__ void scatter_add_kernel(const float* __restrict__ msg,
                                   const int* __restrict__ src,
                                   const int* __restrict__ dst,
                                   float* __restrict__ out, int nedges, int fo) {
  int c4 = fo >> 2;
  int t = blockIdx.x * blockDim.x + threadIdx.x;
  if (t >= nedges * c4) return;
  int e = t / c4;
  int c = (t % c4) * 4;
  const float4 v = *(const float4*)(msg + (size_t)src[e] * fo + c);
  float* o = out + (size_t)dst[e] * fo + c;
  atomicAdd(o + 0, v.x); atomicAdd(o + 1, v.y);
  atomicAdd(o + 2, v.z); atomicAdd(o + 3, v.w);
}

// ------------- BN stats of relu(acc + bias) ---------------------------------
__global__ void bn_stats_kernel(const float* __restrict__ acc,
                                const float* __restrict__ bias,
                                float* __restrict__ stats, int n, int fo) {
  int t = blockIdx.x * blockDim.x + threadIdx.x;
  int c = t % fo;
  int r0 = t / fo;
  int rstride = (gridDim.x * blockDim.x) / fo;
  float b = bias[c];
  float s = 0.f, s2 = 0.f;
  for (int r = r0; r < n; r += rstride) {
    float v = fmaxf(acc[(size_t)r * fo + c] + b, 0.f);
    s += v; s2 += v * v;
  }
  atomicAdd(&stats[c], s);
  atomicAdd(&stats[fo + c], s2);
}

// ------------- normalize (bias+relu re-applied on read) ---------------------
__global__ void bn_norm_kernel(const float* __restrict__ acc,
                               const float* __restrict__ bias,
                               const float* __restrict__ gamma,
                               const float* __restrict__ beta,
                               const float* __restrict__ stats,
                               bf16* __restrict__ out, int n, int fo) {
  int t = blockIdx.x * blockDim.x + threadIdx.x;
  if (t >= n * fo) return;
  int c = t % fo;
  float inv_n = 1.f / (float)n;
  float mu = stats[c] * inv_n;
  float var = stats[fo + c] * inv_n - mu * mu;
  float v = fmaxf(acc[t] + bias[c], 0.f);
  float y = (v - mu) * rsqrtf(var + BN_EPS) * gamma[c] + beta[c];
  out[t] = (bf16)y;
}

// ------------- pooling ------------------------------------------------------
__global__ void pool_count_kernel(const int* __restrict__ batch, float* __restrict__ counts,
                                  int n) {
  int t = blockIdx.x * blockDim.x + threadIdx.x;
  if (t >= n) return;
  atomicAdd(&counts[batch[t]], 1.0f);
}

__global__ void pool_sum_kernel(const bf16* __restrict__ h, const int* __restrict__ batch,
                                float* __restrict__ pooled, int n) {
  int t = blockIdx.x * blockDim.x + threadIdx.x;
  if (t >= n * 256) return;
  int node = t >> 8, c = t & 255;
  atomicAdd(&pooled[batch[node] * 256 + c], (float)h[t]);
}

__global__ void pool_mean_kernel(const float* __restrict__ pooled,
                                 const float* __restrict__ counts,
                                 float* __restrict__ pmean) {
  int t = blockIdx.x * blockDim.x + threadIdx.x;
  if (t >= NGRAPH * 256) return;
  pmean[t] = pooled[t] / fmaxf(counts[t >> 8], 1.0f);
}

// ------------- dense FC (tiny: 64 rows, keep f32) ---------------------------
__global__ void fc_kernel(const float* __restrict__ in, const float* __restrict__ W,
                          const float* __restrict__ b, float* __restrict__ out,
                          int rows, int fin, int fout, int do_relu) {
  int t = blockIdx.x * blockDim.x + threadIdx.x;
  if (t >= rows * fout) return;
  int r = t / fout, j = t % fout;
  float s = b[j];
  const float* irow = in + (size_t)r * fin;
  for (int k = 0; k < fin; ++k) s += irow[k] * W[(size_t)k * fout + j];
  if (do_relu) s = fmaxf(s, 0.f);
  out[t] = s;
}

// ---------------------------------------------------------------------------
static inline int nblk(long long total, int bs) { return (int)((total + bs - 1) / bs); }

extern "C" void kernel_launch(void* const* d_in, const int* in_sizes, int n_in,
                              void* d_out, int out_size, void* d_ws, size_t ws_size,
                              hipStream_t stream) {
  (void)in_sizes; (void)n_in; (void)out_size; (void)ws_size;

  const float* x     = (const float*)d_in[0];
  const int*   ei    = (const int*)d_in[1];      // [2, E] flattened
  const int*   src   = ei;
  const int*   dst   = ei + NEDGES;
  const int*   batch = (const int*)d_in[2];

  // Param leaves in dict-insertion order: 9 x (W, b, gamma, beta), then fc.
  // dims: {actual fi, padded fi (mult. of 32), fo}
  static const int dims[9][3] = {
      {19, 32, 64}, {64, 64, 64}, {64, 64, 128},
      {128, 128, 128}, {128, 128, 128}, {128, 128, 128},
      {128, 128, 256}, {256, 256, 256}, {256, 256, 256}};
  const float *Wl[9], *bl[9], *gl[9], *tl[9];
  int p = 3;
  for (int l = 0; l < 9; ++l) {
    Wl[l] = (const float*)d_in[p + 0];
    bl[l] = (const float*)d_in[p + 1];
    gl[l] = (const float*)d_in[p + 2];
    tl[l] = (const float*)d_in[p + 3];
    p += 4;
  }
  const float* fcW1 = (const float*)d_in[p + 0];
  const float* fcb1 = (const float*)d_in[p + 1];
  const float* fcW2 = (const float*)d_in[p + 2];
  const float* fcb2 = (const float*)d_in[p + 3];

  // ---- carve workspace (256B aligned slices) ----
  char* ws = (char*)d_ws;
  size_t off = 0;
  auto carve = [&](size_t bytes) -> char* {
    char* ptr = ws + off;
    off += (bytes + 255) & ~(size_t)255;
    return ptr;
  };
  bf16*  hA     = (bf16*)carve((size_t)NNODES * 256 * sizeof(bf16));   // 51.2 MB
  bf16*  hB     = (bf16*)carve((size_t)NNODES * 256 * sizeof(bf16));   // 51.2 MB
  float* gemm_o = (float*)carve((size_t)NNODES * 256 * sizeof(float)); // 102.4 MB
  float* accum  = (float*)carve((size_t)NNODES * 256 * sizeof(float)); // 102.4 MB
  bf16*  Wt     = (bf16*)carve((size_t)256 * 256 * sizeof(bf16));
  float* stats  = (float*)carve(512 * sizeof(float));
  float* pooled = (float*)carve((size_t)NGRAPH * 256 * sizeof(float));
  float* counts = (float*)carve((size_t)NGRAPH * sizeof(float));
  float* pmean  = (float*)carve((size_t)NGRAPH * 256 * sizeof(float));
  float* fcbuf  = (float*)carve((size_t)NGRAPH * 256 * sizeof(float));

  // ---- features: mask 39 -> 19 columns, zero-pad to 32, convert to bf16 ----
  feat_mask_kernel<<<nblk((long long)NNODES * 32, 256), 256, 0, stream>>>(x, hA, NNODES);

  bf16* cur = hA;
  bf16* nxt = hB;
  for (int l = 0; l < 9; ++l) {
    const int fi = dims[l][0], fip = dims[l][1], fo = dims[l][2];

    conv_w_kernel<<<nblk((long long)fip * fo, 256), 256, 0, stream>>>(Wl[l], Wt, fi, fip, fo);

    // WMMA GEMM: gemm_o[N, fo] = cur[N, fip] @ W   (16x64 tile per wave)
    dim3 gblock(32, 4);
    dim3 ggrid((6250 + 3) / 4, fo / 64);
    gemm_bf16_wmma_kernel<<<ggrid, gblock, 0, stream>>>(cur, Wt, gemm_o, NNODES, fip, fo);

    // scatter-sum over edges into accum
    hipMemsetAsync(accum, 0, (size_t)NNODES * fo * sizeof(float), stream);
    scatter_add_kernel<<<nblk((long long)NEDGES * (fo >> 2), 256), 256, 0, stream>>>(
        gemm_o, src, dst, accum, NEDGES, fo);

    // BN stats of relu(accum + bias), then normalize -> bf16 for next GEMM
    hipMemsetAsync(stats, 0, 2 * fo * sizeof(float), stream);
    bn_stats_kernel<<<512, 256, 0, stream>>>(accum, bl[l], stats, NNODES, fo);
    bn_norm_kernel<<<nblk((long long)NNODES * fo, 256), 256, 0, stream>>>(
        accum, bl[l], gl[l], tl[l], stats, nxt, NNODES, fo);

    bf16* t = cur; cur = nxt; nxt = t;
  }

  // ---- global mean pool per graph ----
  hipMemsetAsync(pooled, 0, (size_t)NGRAPH * 256 * sizeof(float), stream);
  hipMemsetAsync(counts, 0, (size_t)NGRAPH * sizeof(float), stream);
  pool_count_kernel<<<nblk(NNODES, 256), 256, 0, stream>>>(batch, counts, NNODES);
  pool_sum_kernel<<<nblk((long long)NNODES * 256, 256), 256, 0, stream>>>(cur, batch, pooled, NNODES);
  pool_mean_kernel<<<nblk((long long)NGRAPH * 256, 256), 256, 0, stream>>>(pooled, counts, pmean);

  // ---- FC head ----
  fc_kernel<<<nblk((long long)NGRAPH * 256, 256), 256, 0, stream>>>(
      pmean, fcW1, fcb1, fcbuf, NGRAPH, 256, 256, 1);
  fc_kernel<<<nblk((long long)NGRAPH * NCLASS, 256), 256, 0, stream>>>(
      fcbuf, fcW2, fcb2, (float*)d_out, NGRAPH, 256, NCLASS, 0);
}
